// Lepton_59158879535763
// MI455X (gfx1250) — compile-verified, matrix operand order checked by reading, not run on hardware
//
#include <hip/hip_runtime.h>
#include <hip/hip_bf16.h>
#include <math.h>

// ---------------- model constants ----------------
#define Cc   1024
#define Hh   16
#define Dh   64
#define Ll   8
#define NTt  4
#define Bb   2
#define Nn   1024
#define Mm   (NTt + Nn)        // 1028
#define BMm  (Bb * Mm)         // 2056
#define BNn  (Bb * Nn)         // 2048
#define HDd  2752
#define Vv   32000
#define CSz  64
#define NCh  ((Mm + CSz - 1) / CSz)   // 17
#define PK   8
#define KPEEK (Cc * PK)        // 8192

typedef __attribute__((ext_vector_type(16))) __bf16 v16bf;
typedef __attribute__((ext_vector_type(8)))  float  v8f;

__device__ __forceinline__ __bf16 f2bf(float x) { return (__bf16)x; }
__device__ __forceinline__ float  sigmoidf_(float x) { return 1.f / (1.f + __expf(-x)); }
__device__ __forceinline__ float  geluf_(float x) { return 0.5f * x * (1.f + erff(x * 0.70710678118f)); }

// Pack 4 f32 -> 4 bf16 and store as one 8-byte LDS write (ds_store_b64).
__device__ __forceinline__ void st4bf(__bf16* dst, float4 v) {
  union { uint2 u; __bf16 e[4]; } p;
  p.e[0] = f2bf(v.x); p.e[1] = f2bf(v.y); p.e[2] = f2bf(v.z); p.e[3] = f2bf(v.w);
  *(uint2*)dst = p.u;
}

// ---------------- WMMA fragment loader ----------------
// 16x32 bf16 operand tile, storage [row][k], stride in elements (multiple of 8
// so each half-fragment is a 16B-aligned ds_load_b128).
// lane L: row = L&15 ; K pattern {b0..b0+7, 16+b0..16+b0+7}, b0 = (L>>4)*8.
__device__ __forceinline__ v16bf frag_ld(const __bf16* base, int stride) {
  const int lane = threadIdx.x & 31;
  const int b0 = (lane >> 4) << 3;
  const __bf16* p = base + (lane & 15) * stride + b0;
  union { v16bf v; uint4 q[2]; } f;
  f.q[0] = *(const uint4*)(p);
  f.q[1] = *(const uint4*)(p + 16);
  return f.v;
}

// ---------------- generic WMMA GEMM (software-pipelined) ----------------
// D[M,N](ldd) = act( A[M,K](lda) @ W[N,K](ldw)^T + bias )  [optionally +=]
// Tile: 64(M) x 128(N), 8 waves, 4 accumulators/wave. K%32==0, K%4==0.
// Global loads for slab k+1 are issued before the WMMAs of slab k so that
// HBM latency stays outstanding (single LOADcnt drain per slab).
// MASK=false is the block-uniform fast path (tile fully in-bounds).
#define GF_ACC  1
#define GF_BIAS 2
#define GF_GELU 4

template <bool MASK>
__device__ __forceinline__ void gemm_mainloop(
    const float* __restrict__ A, int lda,
    const float* __restrict__ W, int ldw,
    __bf16* As, __bf16* Bs,
    int bm, int bn, int M, int N, int K,
    int rr, int c4, int wm, int wn, v8f acc[4])
{
  float4 ar[2], br[4];
#pragma unroll
  for (int it = 0; it < 2; ++it) {
    int gm = bm + rr + it * 32;
    int gmc = MASK ? (gm < M ? gm : (M - 1)) : gm;
    ar[it] = *(const float4*)(A + (size_t)gmc * lda + c4);
  }
#pragma unroll
  for (int it = 0; it < 4; ++it) {
    int gn = bn + rr + it * 32;
    int gnc = MASK ? (gn < N ? gn : (N - 1)) : gn;
    br[it] = *(const float4*)(W + (size_t)gnc * ldw + c4);
  }

  for (int kk = 0; kk < K; kk += 32) {
    // mask + convert + store current slab to LDS
#pragma unroll
    for (int it = 0; it < 2; ++it) {
      float4 v = ar[it];
      if (MASK && (bm + rr + it * 32 >= M)) v = make_float4(0.f, 0.f, 0.f, 0.f);
      st4bf(As + (rr + it * 32) * 40 + c4, v);
    }
#pragma unroll
    for (int it = 0; it < 4; ++it) {
      float4 v = br[it];
      if (MASK && (bn + rr + it * 32 >= N)) v = make_float4(0.f, 0.f, 0.f, 0.f);
      st4bf(Bs + (rr + it * 32) * 40 + c4, v);
    }
    __syncthreads();

    // issue next slab's global loads; consumed only next iteration
    if (kk + 32 < K) {
      const int k2 = kk + 32;
#pragma unroll
      for (int it = 0; it < 2; ++it) {
        int gm = bm + rr + it * 32;
        int gmc = MASK ? (gm < M ? gm : (M - 1)) : gm;
        ar[it] = *(const float4*)(A + (size_t)gmc * lda + k2 + c4);
      }
#pragma unroll
      for (int it = 0; it < 4; ++it) {
        int gn = bn + rr + it * 32;
        int gnc = MASK ? (gn < N ? gn : (N - 1)) : gn;
        br[it] = *(const float4*)(W + (size_t)gnc * ldw + k2 + c4);
      }
    }

    // matrix work for current slab
    v16bf af = frag_ld(As + wm * 40, 40);
#pragma unroll
    for (int t = 0; t < 4; ++t) {
      v16bf bf = frag_ld(Bs + (wn + t * 16) * 40, 40);
      acc[t] = __builtin_amdgcn_wmma_f32_16x16x32_bf16(
          false, af, false, bf, (short)0, acc[t], false, false);
    }
    __syncthreads();
  }
}

__global__ __launch_bounds__(256) void gemm_wmma(
    const float* __restrict__ A, int lda,
    const float* __restrict__ W, int ldw,
    const float* __restrict__ bias,
    float* __restrict__ D, int ldd,
    int M, int N, int K, int flags)
{
  __shared__ __align__(16) __bf16 As[64 * 40];
  __shared__ __align__(16) __bf16 Bs[128 * 40];
  const int tid = threadIdx.x;
  const int lane = tid & 31, wave = tid >> 5;
  const int bm = blockIdx.y * 64, bn = blockIdx.x * 128;
  const int wm = (wave & 3) * 16;          // wave row tile
  const int wn = (wave >> 2) * 64;         // wave col base (4 tiles)
  const int rr = tid >> 3;                 // staging row base 0..31
  const int c4 = (tid & 7) * 4;            // staging k offset 0..28
  v8f acc[4] = { (v8f){}, (v8f){}, (v8f){}, (v8f){} };

  const bool full = (bm + 64 <= M) && (bn + 128 <= N);
  if (full)
    gemm_mainloop<false>(A, lda, W, ldw, As, Bs, bm, bn, M, N, K, rr, c4, wm, wn, acc);
  else
    gemm_mainloop<true>(A, lda, W, ldw, As, Bs, bm, bn, M, N, K, rr, c4, wm, wn, acc);

  // epilogue
  const int n0 = lane & 15, mh = (lane >> 4) * 8;
  if (full) {
#pragma unroll
    for (int t = 0; t < 4; ++t) {
      int gn = bn + wn + t * 16 + n0;
      float bv = (flags & GF_BIAS) ? bias[gn] : 0.f;
#pragma unroll
      for (int r = 0; r < 8; ++r) {
        int gm = bm + wm + mh + r;
        float v = acc[t][r] + bv;
        if (flags & GF_GELU) v = geluf_(v);
        size_t di = (size_t)gm * ldd + gn;
        if (flags & GF_ACC) D[di] += v; else D[di] = v;
      }
    }
  } else {
#pragma unroll
    for (int t = 0; t < 4; ++t) {
      int gn = bn + wn + t * 16 + n0;
      if (gn >= N) continue;
      float bv = (flags & GF_BIAS) ? bias[gn] : 0.f;
#pragma unroll
      for (int r = 0; r < 8; ++r) {
        int gm = bm + wm + mh + r;
        if (gm < M) {
          float v = acc[t][r] + bv;
          if (flags & GF_GELU) v = geluf_(v);
          size_t di = (size_t)gm * ldd + gn;
          if (flags & GF_ACC) D[di] += v; else D[di] = v;
        }
      }
    }
  }
}

// ---------------- chunked attention (WMMA) ----------------
// one block per (b, head, chunk); qkv rows are (b*M+m), cols = s*C + h*64 + d
__global__ __launch_bounds__(128) void attn_kernel(
    const float* __restrict__ qkv,
    const float* __restrict__ csk, const float* __restrict__ csv,
    float* __restrict__ outb)
{
  const int bid = blockIdx.x;
  const int ci = bid % NCh;
  const int hh = (bid / NCh) % Hh;
  const int b  = bid / (NCh * Hh);
  __shared__ __align__(16) __bf16 Qs[64 * 72];
  __shared__ __align__(16) __bf16 Ks[64 * 72];
  __shared__ __align__(16) __bf16 Vt[64 * 72];   // transposed: [d][key]
  __shared__ __align__(16) __bf16 Ps[64 * 72];
  __shared__ float Sf[64 * 65];
  __shared__ float crossv[64], cskl[64], csvl[64];
  const int tid = threadIdx.x, lane = tid & 31, wave = tid >> 5;
  const float scale = 0.125f;   // D^-0.5

#pragma unroll
  for (int it = 0; it < 8; ++it) {        // 64x64 = 1024 float4 chunks
    int idx = tid + it * 128;
    int j = idx >> 4, d4 = (idx & 15) * 4;
    int m = ci * CSz + j;
    float4 q4 = make_float4(0.f, 0.f, 0.f, 0.f), k4 = q4, v4 = q4;
    if (m < Mm) {
      size_t base = (size_t)(b * Mm + m) * (3 * Cc) + hh * Dh + d4;
      q4 = *(const float4*)(qkv + base);
      k4 = *(const float4*)(qkv + base + Cc);
      v4 = *(const float4*)(qkv + base + 2 * Cc);
    }
    st4bf(Qs + j * 72 + d4, q4);
    st4bf(Ks + j * 72 + d4, k4);
    Vt[(d4 + 0) * 72 + j] = f2bf(v4.x);
    Vt[(d4 + 1) * 72 + j] = f2bf(v4.y);
    Vt[(d4 + 2) * 72 + j] = f2bf(v4.z);
    Vt[(d4 + 3) * 72 + j] = f2bf(v4.w);
  }
  if (tid < 64) {
    size_t cb = (size_t)((b * Hh + hh) * NCh + ci) * 64 + tid;
    cskl[tid] = csk[cb]; csvl[tid] = csv[cb];
  }
  __syncthreads();

  // S = scale * Q K^T
  {
    const int wm = wave * 16;
    v8f acc[4] = { (v8f){}, (v8f){}, (v8f){}, (v8f){} };
#pragma unroll
    for (int kk = 0; kk < 64; kk += 32) {
      v16bf a = frag_ld(Qs + wm * 72 + kk, 72);
#pragma unroll
      for (int t = 0; t < 4; ++t) {
        v16bf bf = frag_ld(Ks + t * 16 * 72 + kk, 72);
        acc[t] = __builtin_amdgcn_wmma_f32_16x16x32_bf16(
            false, a, false, bf, (short)0, acc[t], false, false);
      }
    }
    const int n0 = lane & 15, mh = (lane >> 4) * 8;
#pragma unroll
    for (int t = 0; t < 4; ++t)
#pragma unroll
      for (int r = 0; r < 8; ++r)
        Sf[(wm + mh + r) * 65 + t * 16 + n0] = acc[t][r] * scale;
  }
  __syncthreads();

  // masked softmax per row + cross-chunk sigmoid gate
  if (tid < 64) {
    const int j = tid;
    float mx = -1e30f;
    for (int n = 0; n < 64; ++n) {
      float s = (n > j) ? -65000.f : Sf[j * 65 + n];
      Sf[j * 65 + n] = s;
      mx = fmaxf(mx, s);
    }
    float sum = 0.f;
    for (int n = 0; n < 64; ++n) {
      float e = __expf(Sf[j * 65 + n] - mx);
      Sf[j * 65 + n] = e; sum += e;
    }
    float inv = 1.f / sum;
    for (int n = 0; n < 64; ++n) Ps[j * 72 + n] = f2bf(Sf[j * 65 + n] * inv);
    float dot = 0.f;
    for (int d = 0; d < 64; ++d) dot += (float)Qs[j * 72 + d] * cskl[d];
    crossv[j] = sigmoidf_(dot * scale);
  }
  __syncthreads();

  // O = P V  (+ cross * csv * 0.5)
  {
    const int wm = wave * 16;
    v8f acc[4] = { (v8f){}, (v8f){}, (v8f){}, (v8f){} };
#pragma unroll
    for (int kk = 0; kk < 64; kk += 32) {
      v16bf a = frag_ld(Ps + wm * 72 + kk, 72);
#pragma unroll
      for (int t = 0; t < 4; ++t) {
        v16bf bf = frag_ld(Vt + t * 16 * 72 + kk, 72);
        acc[t] = __builtin_amdgcn_wmma_f32_16x16x32_bf16(
            false, a, false, bf, (short)0, acc[t], false, false);
      }
    }
    const int n0 = lane & 15, mh = (lane >> 4) * 8;
#pragma unroll
    for (int t = 0; t < 4; ++t)
#pragma unroll
      for (int r = 0; r < 8; ++r) {
        int j = wm + mh + r;
        int m = ci * CSz + j;
        if (m < Mm) {
          int d = t * 16 + n0;
          float v = acc[t][r] + crossv[j] * csvl[d] * 0.5f;
          outb[(size_t)(b * Mm + m) * Cc + hh * Dh + d] = v;
        }
      }
  }
}

// ---------------- exclusive cumulative chunk means of K,V ----------------
__global__ void cskv_kernel(const float* __restrict__ qkv,
                            float* __restrict__ csk, float* __restrict__ csv)
{
  const int d = threadIdx.x;                 // 64 threads
  const int hh = blockIdx.x % Hh, b = blockIdx.x / Hh;
  float rk = 0.f, rv = 0.f;
  for (int ci = 0; ci < NCh; ++ci) {
    size_t ob = (size_t)((b * Hh + hh) * NCh + ci) * 64 + d;
    csk[ob] = rk; csv[ob] = rv;
    float sk = 0.f, sv = 0.f;
    for (int j = 0; j < CSz; ++j) {
      int m = ci * CSz + j;
      if (m >= Mm) break;
      size_t base = (size_t)(b * Mm + m) * (3 * Cc) + hh * Dh + d;
      sk += qkv[base + Cc]; sv += qkv[base + 2 * Cc];
    }
    rk += sk * (1.f / 64.f); rv += sv * (1.f / 64.f);
  }
}

// ---------------- elementwise / small kernels ----------------
__global__ __launch_bounds__(128) void rmsnorm_kernel(
    const float* __restrict__ in, const float* __restrict__ w,
    float* __restrict__ out, int Bx, int Mi, int off, int No)
{
  const int row = blockIdx.x;                   // Bx*No rows
  const int b = row / No, r = row % No;
  const float* x = in + (size_t)(b * Mi + off + r) * Cc;
  float* y = out + (size_t)row * Cc;
  __shared__ float red[128];
  float s = 0.f;
  for (int c = threadIdx.x; c < Cc; c += 128) { float v = x[c]; s += v * v; }
  red[threadIdx.x] = s; __syncthreads();
  for (int st = 64; st > 0; st >>= 1) {
    if (threadIdx.x < st) red[threadIdx.x] += red[threadIdx.x + st];
    __syncthreads();
  }
  float rs = rsqrtf(red[0] * (1.f / Cc) + 1e-6f);
  for (int c = threadIdx.x; c < Cc; c += 128) y[c] = x[c] * rs * w[c];
}

__global__ void emb_gather_kernel(const int* __restrict__ x,
                                  const float* __restrict__ emb,
                                  float* __restrict__ h)
{
  size_t i = (size_t)blockIdx.x * blockDim.x + threadIdx.x;
  if (i >= (size_t)BNn * Cc) return;
  int c = i % Cc; int t = i / Cc;
  int b = t / Nn, n = t % Nn;
  int tok = x[b * Nn + n];
  h[(size_t)(b * Mm + NTt + n) * Cc + c] = emb[(size_t)tok * Cc + c];
}

__global__ void mean_kernel(const float* __restrict__ h, float* __restrict__ mb)
{
  int i = blockIdx.x * blockDim.x + threadIdx.x;
  if (i >= Bb * Cc) return;
  int b = i / Cc, c = i % Cc;
  float s = 0.f;
  for (int n = 0; n < Nn; ++n) s += h[(size_t)(b * Mm + NTt + n) * Cc + c];
  mb[i] = s * (1.f / Nn);
}

__global__ void cond_kernel(const float* __restrict__ mb,
                            const float* __restrict__ Wp, float* __restrict__ cond)
{
  int i = blockIdx.x * blockDim.x + threadIdx.x;
  if (i >= Bb * Cc) return;
  int b = i / Cc, co = i % Cc;
  float s = 0.f;
  for (int c = 0; c < Cc; ++c) s += mb[b * Cc + c] * Wp[(size_t)co * Cc + c];
  cond[i] = s;
}

__global__ void thoughts_kernel(const float* __restrict__ te,
                                const float* __restrict__ cond, float* __restrict__ h)
{
  int i = blockIdx.x * blockDim.x + threadIdx.x;
  if (i >= Bb * NTt * Cc) return;
  int c = i % Cc; int t = (i / Cc) % NTt; int b = i / (Cc * NTt);
  h[(size_t)(b * Mm + t) * Cc + c] = te[t * Cc + c] + cond[b * Cc + c];
}

__global__ __launch_bounds__(128) void memread_kernel(
    float* __restrict__ h, const float* __restrict__ rq,
    const float* __restrict__ mem, const float* __restrict__ gw)
{
  int t = blockIdx.x * blockDim.x + threadIdx.x;
  if (t >= BMm) return;
  const float* q = rq + (size_t)t * Cc;
  float sc[8]; float mx = -1e30f;
  for (int s = 0; s < 8; ++s) {
    float d = 0.f;
    for (int c = 0; c < Cc; ++c) d += q[c] * mem[s * Cc + c];
    sc[s] = d * 0.03125f;                     // / sqrt(C)
    mx = fmaxf(mx, sc[s]);
  }
  float sum = 0.f;
  for (int s = 0; s < 8; ++s) { sc[s] = __expf(sc[s] - mx); sum += sc[s]; }
  float inv = 1.f / sum;
  float* hr = h + (size_t)t * Cc;
  float g = 0.f;
  for (int c = 0; c < Cc; ++c) g += hr[c] * gw[c];
  g = sigmoidf_(g);
  for (int c = 0; c < Cc; ++c) {
    float mv = 0.f;
    for (int s = 0; s < 8; ++s) mv += sc[s] * inv * mem[s * Cc + c];
    hr[c] += g * mv;
  }
}

__global__ void peek_gather_kernel(const float* __restrict__ h, float* __restrict__ pk)
{
  const size_t total = (size_t)BMm * KPEEK;
  for (size_t i = (size_t)blockIdx.x * blockDim.x + threadIdx.x; i < total;
       i += (size_t)gridDim.x * blockDim.x) {
    int rem = i % KPEEK; size_t t = i / KPEEK;
    int c = rem >> 3, p = rem & 7;
    int b = (int)(t / Mm), m = (int)(t % Mm);
    int mp = m + 1 + p;
    pk[i] = (mp < Mm) ? h[(size_t)(b * Mm + mp) * Cc + c] : 0.f;
  }
}

__global__ void peek_apply_kernel(float* __restrict__ h,
                                  const float* __restrict__ G,
                                  const float* __restrict__ pc)
{
  size_t i = (size_t)blockIdx.x * blockDim.x + threadIdx.x;
  if (i >= (size_t)BMm * Cc) return;
  h[i] += sigmoidf_(G[i]) * pc[i] * 0.5f;
}

__global__ void router_kernel(const float* __restrict__ rh,
                              const float* __restrict__ w2,
                              const float* __restrict__ b2, float* __restrict__ rw)
{
  int t = blockIdx.x * blockDim.x + threadIdx.x;
  if (t >= BMm) return;
  const float* x = rh + (size_t)t * (Cc / 4);
  float l[3], mx = -1e30f;
  for (int j = 0; j < 3; ++j) {
    float s = b2[j];
    for (int c = 0; c < Cc / 4; ++c) s += x[c] * w2[j * (Cc / 4) + c];
    l[j] = s; mx = fmaxf(mx, s);
  }
  float sum = 0.f;
  for (int j = 0; j < 3; ++j) { l[j] = __expf(l[j] - mx); sum += l[j]; }
  for (int j = 0; j < 3; ++j) rw[t * 3 + j] = l[j] / sum;
}

__global__ void silu_mul_kernel(float* __restrict__ h1, const float* __restrict__ h3)
{
  size_t i = (size_t)blockIdx.x * blockDim.x + threadIdx.x;
  if (i >= (size_t)BMm * HDd) return;
  float a = h1[i];
  h1[i] = a * sigmoidf_(a) * h3[i];
}

__global__ void combine_kernel(const float* __restrict__ outs,
                               const float* __restrict__ rw, float* __restrict__ dst)
{
  size_t i = (size_t)blockIdx.x * blockDim.x + threadIdx.x;
  if (i >= (size_t)BMm * Cc) return;
  size_t t = i / Cc;
  const size_t stride = (size_t)BMm * Cc;
  dst[i] = outs[i] * rw[t * 3 + 0] + outs[i + stride] * rw[t * 3 + 1] +
           outs[i + 2 * stride] * rw[t * 3 + 2];
}

__global__ void conf_head_kernel(const float* __restrict__ ch,
                                 const float* __restrict__ w2,
                                 const float* __restrict__ b2, float* __restrict__ out)
{
  int t = blockIdx.x * blockDim.x + threadIdx.x;
  if (t >= BNn) return;
  const float* x = ch + (size_t)t * (Cc / 4);
  for (int j = 0; j < 4; ++j) {
    float s = b2[j];
    for (int c = 0; c < Cc / 4; ++c) s += x[c] * w2[j * (Cc / 4) + c];
    out[t * 4 + j] = sigmoidf_(s);
  }
}

__global__ void unc_head_kernel(const float* __restrict__ uh,
                                const float* __restrict__ w2,
                                const float* __restrict__ b2, float* __restrict__ out)
{
  int t = blockIdx.x * blockDim.x + threadIdx.x;
  if (t >= BNn) return;
  const float* x = uh + (size_t)t * (Cc / 2);
  float s = b2[0];
  for (int c = 0; c < Cc / 2; ++c) s += x[c] * w2[c];
  out[t] = sigmoidf_(s);
}

// ---------------- host-side orchestration ----------------
static inline void gemm(hipStream_t st, const float* A, int lda, const float* W, int ldw,
                        const float* bias, float* D, int ldd, int M, int N, int K, int flags)
{
  dim3 g((N + 127) / 128, (M + 63) / 64);
  gemm_wmma<<<g, 256, 0, st>>>(A, lda, W, ldw, bias, D, ldd, M, N, K, flags);
}
static inline int cdiv(size_t a, int b) { return (int)((a + b - 1) / b); }

extern "C" void kernel_launch(void* const* d_in, const int* in_sizes, int n_in,
                              void* d_out, int out_size, void* d_ws, size_t ws_size,
                              hipStream_t stream) {
  (void)in_sizes; (void)n_in; (void)out_size; (void)ws_size;
  const int*   x        = (const int*)d_in[0];
  const float* emb_w    = (const float*)d_in[1];
  const float* te       = (const float*)d_in[2];
  const float* tproj    = (const float*)d_in[3];
  const float* mem      = (const float*)d_in[4];
  const float* mem_rq_w = (const float*)d_in[5];
  const float* mem_gw   = (const float*)d_in[6];
  const float* pk_cw    = (const float*)d_in[7];
  const float* pk_gw    = (const float*)d_in[8];
  const float* r_w1     = (const float*)d_in[9];
  const float* r_b1     = (const float*)d_in[10];
  const float* r_w2     = (const float*)d_in[11];
  const float* r_b2     = (const float*)d_in[12];
  const float* n1_w     = (const float*)d_in[13];
  const float* qkv_w    = (const float*)d_in[14];
  const float* ao_w     = (const float*)d_in[15];
  const float* n2_w     = (const float*)d_in[16];
  const float* mlp_w1   = (const float*)d_in[17];
  const float* mlp_w2   = (const float*)d_in[18];
  const float* mlp_w3   = (const float*)d_in[19];
  const float* dn_w     = (const float*)d_in[20];
  const float* no_w     = (const float*)d_in[21];
  const float* spec_w   = (const float*)d_in[22];
  const float* c_w1     = (const float*)d_in[23];
  const float* c_b1     = (const float*)d_in[24];
  const float* c_w2     = (const float*)d_in[25];
  const float* c_b2     = (const float*)d_in[26];
  const float* u_w1     = (const float*)d_in[27];
  const float* u_b1     = (const float*)d_in[28];
  const float* u_w2     = (const float*)d_in[29];
  const float* u_b2     = (const float*)d_in[30];

  // workspace layout (floats)
  float* w = (float*)d_ws;
  size_t o = 0;
  float* h      = w + o; o += (size_t)BMm * Cc;
  float* tmp1   = w + o; o += (size_t)BMm * Cc;          // rmsnorm out / pc
  float* qkvb   = w + o; o += (size_t)BMm * 3 * Cc;      // also rq
  float* attnb  = w + o; o += (size_t)BMm * Cc;          // also peek gate G
  float* h1     = w + o; o += (size_t)BMm * HDd;         // peeks aliases [h1 ..)
  float* h3     = w + o; o += (size_t)BMm * HDd;
  float* outs   = w + o; o += (size_t)3 * BMm * Cc;
  float* rw     = w + o; o += (size_t)BMm * 4;
  float* condb  = w + o; o += (size_t)Bb * Cc;
  float* meanb  = w + o; o += (size_t)Bb * Cc;
  float* rh     = w + o; o += (size_t)BMm * (Cc / 4);
  float* csk    = w + o; o += (size_t)Bb * Hh * NCh * 64;
  float* csv    = w + o; o += (size_t)Bb * Hh * NCh * 64;
  float* hout   = w + o; o += (size_t)BNn * Cc;
  float* confh  = w + o; o += (size_t)BNn * (Cc / 4);
  float* unch   = w + o; o += (size_t)BNn * (Cc / 2);
  float* peeks  = h1;                                    // lifetime-disjoint alias
  float* rq     = qkvb;
  float* G      = attnb;
  float* pc     = tmp1;

  // 1) embedding gather + thoughts
  emb_gather_kernel<<<cdiv((size_t)BNn * Cc, 256), 256, 0, stream>>>(x, emb_w, h);
  mean_kernel<<<cdiv(Bb * Cc, 256), 256, 0, stream>>>(h, meanb);
  cond_kernel<<<cdiv(Bb * Cc, 256), 256, 0, stream>>>(meanb, tproj, condb);
  thoughts_kernel<<<cdiv(Bb * NTt * Cc, 256), 256, 0, stream>>>(te, condb, h);

  // 2) memory read
  gemm(stream, h, Cc, mem_rq_w, Cc, nullptr, rq, Cc, BMm, Cc, Cc, 0);
  memread_kernel<<<cdiv(BMm, 128), 128, 0, stream>>>(h, rq, mem, mem_gw);

  // 3) future peek
  peek_gather_kernel<<<4096, 256, 0, stream>>>(h, peeks);
  gemm(stream, peeks, KPEEK, pk_cw, KPEEK, nullptr, pc, Cc, BMm, Cc, KPEEK, 0);
  gemm(stream, h, Cc, pk_gw, 2 * Cc, nullptr, G, Cc, BMm, Cc, Cc, 0);
  gemm(stream, pc, Cc, pk_gw + Cc, 2 * Cc, nullptr, G, Cc, BMm, Cc, Cc, GF_ACC);
  peek_apply_kernel<<<cdiv((size_t)BMm * Cc, 256), 256, 0, stream>>>(h, G, pc);

  // 4) router weights
  gemm(stream, h, Cc, r_w1, Cc, r_b1, rh, Cc / 4, BMm, Cc / 4, Cc, GF_BIAS | GF_GELU);
  router_kernel<<<cdiv(BMm, 128), 128, 0, stream>>>(rh, r_w2, r_b2, rw);

  // 5) transformer layers
  int depth_slot = 0;
  for (int i = 0; i < Ll; ++i) {
    rmsnorm_kernel<<<BMm, 128, 0, stream>>>(h, n1_w + (size_t)i * Cc, tmp1, Bb, Mm, 0, Mm);
    gemm(stream, tmp1, Cc, qkv_w + (size_t)i * 3 * Cc * Cc, Cc, nullptr,
         qkvb, 3 * Cc, BMm, 3 * Cc, Cc, 0);
    cskv_kernel<<<Bb * Hh, 64, 0, stream>>>(qkvb, csk, csv);
    attn_kernel<<<Bb * Hh * NCh, 128, 0, stream>>>(qkvb, csk, csv, attnb);
    gemm(stream, attnb, Cc, ao_w + (size_t)i * Cc * Cc, Cc, nullptr,
         h, Cc, BMm, Cc, Cc, GF_ACC);

    rmsnorm_kernel<<<BMm, 128, 0, stream>>>(h, n2_w + (size_t)i * Cc, tmp1, Bb, Mm, 0, Mm);
    gemm(stream, tmp1, Cc, mlp_w1 + (size_t)i * HDd * Cc, Cc, nullptr,
         h1, HDd, BMm, HDd, Cc, 0);
    gemm(stream, tmp1, Cc, mlp_w3 + (size_t)i * HDd * Cc, Cc, nullptr,
         h3, HDd, BMm, HDd, Cc, 0);
    silu_mul_kernel<<<cdiv((size_t)BMm * HDd, 256), 256, 0, stream>>>(h1, h3);
    gemm(stream, h1, HDd, mlp_w2 + (size_t)i * Cc * HDd, HDd, nullptr,
         h, Cc, BMm, Cc, HDd, GF_ACC);

    int dep = i + 1;
    if (dep == 2 || dep == 4 || dep == 8) {
      rmsnorm_kernel<<<BMm, 128, 0, stream>>>(h, dn_w + (size_t)depth_slot * Cc,
                                              outs + (size_t)depth_slot * BMm * Cc,
                                              Bb, Mm, 0, Mm);
      ++depth_slot;
    }
  }

  // 6) depth mixture + final norm
  combine_kernel<<<cdiv((size_t)BMm * Cc, 256), 256, 0, stream>>>(outs, rw, tmp1);
  rmsnorm_kernel<<<BNn, 128, 0, stream>>>(tmp1, no_w, hout, Bb, Mm, NTt, Nn);

  // 7) heads
  float* out_f = (float*)d_out;
  const size_t LOG = (size_t)BNn * Vv;
  gemm(stream, hout, Cc, emb_w, Cc, nullptr, out_f, Vv, BNn, Vv, Cc, 0);
  for (int s = 0; s < 4; ++s)
    gemm(stream, hout, Cc, spec_w + (size_t)s * Vv * Cc, Cc, nullptr,
         out_f + LOG + (size_t)s * LOG, Vv, BNn, Vv, Cc, 0);
  gemm(stream, hout, Cc, c_w1, Cc, c_b1, confh, Cc / 4, BNn, Cc / 4, Cc, GF_BIAS | GF_GELU);
  conf_head_kernel<<<cdiv(BNn, 128), 128, 0, stream>>>(confh, c_w2, c_b2, out_f + 5 * LOG);
  gemm(stream, hout, Cc, u_w1, Cc, u_b1, unch, Cc / 2, BNn, Cc / 2, Cc, GF_BIAS | GF_GELU);
  unc_head_kernel<<<cdiv(BNn, 128), 128, 0, stream>>>(unch, u_w2, u_b2,
                                                      out_f + 5 * LOG + (size_t)BNn * 4);
}
